// MmdLoss_31748398252265
// MI455X (gfx1250) — matrix operand order repaired and checked by reading, not compiled, not run on hardware
//
#include <hip/hip_runtime.h>
#include <hip/hip_bf16.h>
#include <math.h>

#define BHALF 4096
#define DIM   1024
#define NTOT  8192

#define MSB  128                 // 8192 / 64 superblocks per side
#define NSB  (MSB * (MSB + 1) / 2)   // 8256 upper-triangular superblocks

typedef float v2f __attribute__((ext_vector_type(2)));
typedef float v8f __attribute__((ext_vector_type(8)));
typedef __bf16 v4bf  __attribute__((ext_vector_type(4)));
typedef __bf16 v8bf  __attribute__((ext_vector_type(8)));
typedef __bf16 v16bf __attribute__((ext_vector_type(16)));

// workspace layout
// float region (first 1 MB):
#define WS_SQ       0        // 8192 row squared norms (fp32, exact)
#define WS_COLPART  8192     // 8 chunks x 1024 partial column sums
#define WS_SCAL     16384    // [0..4] = 1/(bw*2^q)
#define WS_PART     16512    // up to 2*NSB per-wave signed partials
// byte region:
#define WS_HI_BYTES (1ull << 20)                    // 8192x1024 bf16 hi (16 MB)
#define WS_LO_BYTES ((1ull << 20) + (16ull << 20))  // bf16 lo (16 MB)
#define WS_NEED     (33ull * 1024 * 1024)

__device__ __forceinline__ float waveReduce(float v) {
#pragma unroll
  for (int off = 16; off > 0; off >>= 1) v += __shfl_xor(v, off, 32);
  return v;
}

// decode t -> (I,J) with I<=J over the upper triangle of an M x M grid
__device__ __forceinline__ void triDecode(int t, int M, int& I, int& J) {
  int i = 0;
  while (t >= M - i) { t -= M - i; ++i; }
  I = i; J = i + t;
}

// ---------------- stats pass (shared by both paths) ----------------

__global__ __launch_bounds__(256) void k_row_norms(const float* __restrict__ s,
                                                   const float* __restrict__ t,
                                                   float* __restrict__ ws) {
  const int wave = threadIdx.x >> 5, lane = threadIdx.x & 31;
  const int row = blockIdx.x * 8 + wave;
  const float* p = (row < BHALF) ? (s + (size_t)row * DIM)
                                 : (t + (size_t)(row - BHALF) * DIM);
  float acc = 0.f;
  for (int k = lane * 4; k < DIM; k += 128) {
    float4 v = *(const float4*)(p + k);
    acc += v.x * v.x + v.y * v.y + v.z * v.z + v.w * v.w;
  }
  acc = waveReduce(acc);
  if (lane == 0) ws[WS_SQ + row] = acc;
}

__global__ __launch_bounds__(256) void k_colpart(const float* __restrict__ s,
                                                 const float* __restrict__ t,
                                                 float* __restrict__ ws) {
  const int tid = blockIdx.x * 256 + threadIdx.x;   // 0..8191
  const int d = tid & (DIM - 1);
  const int chunk = tid >> 10;                      // 0..7 (1024 rows each)
  const int r0 = chunk * 1024;
  float acc = 0.f;
  for (int i = 0; i < 1024; ++i) {
    const int row = r0 + i;
    const float* p = (row < BHALF) ? (s + (size_t)row * DIM)
                                   : (t + (size_t)(row - BHALF) * DIM);
    acc += p[d];
  }
  ws[WS_COLPART + chunk * DIM + d] = acc;
}

// sum(l2) = 2n*sum(sq) - 2*||colsum||^2  (clamp-free closed form)
__global__ __launch_bounds__(256) void k_scalars(float* __restrict__ ws) {
  __shared__ float sm[256];
  const int t = threadIdx.x;
  float ssum = 0.f;
  for (int i = t; i < NTOT; i += 256) ssum += ws[WS_SQ + i];
  float cnorm = 0.f;
  for (int d = t; d < DIM; d += 256) {
    float cs = 0.f;
#pragma unroll
    for (int c = 0; c < 8; ++c) cs += ws[WS_COLPART + c * DIM + d];
    cnorm += cs * cs;
  }
  sm[t] = ssum; __syncthreads();
  for (int o = 128; o > 0; o >>= 1) { if (t < o) sm[t] += sm[t + o]; __syncthreads(); }
  const float sumSq = sm[0]; __syncthreads();
  sm[t] = cnorm; __syncthreads();
  for (int o = 128; o > 0; o >>= 1) { if (t < o) sm[t] += sm[t + o]; __syncthreads(); }
  if (t == 0) {
    const float sumL2 = 2.f * (float)NTOT * sumSq - 2.f * sm[0];
    float bw = sumL2 / ((float)NTOT * (float)NTOT - (float)NTOT);
    bw *= 0.25f;  // / KERNEL_MUL^(KERNEL_NUM/2)
    float m = 1.f;
    for (int q = 0; q < 5; ++q) { ws[WS_SCAL + q] = 1.f / (bw * m); m *= 2.f; }
  }
}

// ---------------- fast path: split-bf16 (Markidis 3-term) ----------------

__global__ __launch_bounds__(256) void k_convert(const float* __restrict__ s,
                                                 const float* __restrict__ t,
                                                 __bf16* __restrict__ hiA,
                                                 __bf16* __restrict__ loA) {
  const size_t base = ((size_t)blockIdx.x * 256 + threadIdx.x) * 4;
  const size_t row = base >> 10;
  const size_t col = base & (DIM - 1);
  const float* p = (row < BHALF) ? (s + row * DIM + col)
                                 : (t + (row - BHALF) * DIM + col);
  float4 v = *(const float4*)p;
  float f[4] = {v.x, v.y, v.z, v.w};
  v4bf h, l;
#pragma unroll
  for (int i = 0; i < 4; ++i) {
    __bf16 hb = (__bf16)f[i];
    h[i] = hb;
    l[i] = (__bf16)(f[i] - (float)hb);
  }
  *(v4bf*)(hiA + base) = h;
  *(v4bf*)(loA + base) = l;
}

// 2 waves per upper-triangular 64x64 superblock; each wave does a 64x32 half.
// Off-diagonal superblocks weighted 2x (Gram + sign symmetry) -> half the GEMM.
__global__ __launch_bounds__(64) void k_mmd_gemm_bf16(const __bf16* __restrict__ H,
                                                      const __bf16* __restrict__ L,
                                                      float* __restrict__ ws) {
  const int lane = threadIdx.x & 31;   // wave32
  const int wave = threadIdx.x >> 5;   // 0..1
  const int ln = lane & 15, lh = lane >> 4;

  int I, J;
  triDecode(blockIdx.x, MSB, I, J);
  const int rowBase = I * 64;
  const int colBase = J * 64 + wave * 32;
  const float wgt = (I == J) ? 1.f : 2.f;

  // A 16x32 bf16 layout (ISA 7.12.2): lanes0-15 K=[0..7],[16..23]; lanes16-31 +8
  const int aof0 = 8 * lh;
  const int aof1 = 16 + 8 * lh;
  // B 32x16 layout: same K order as A (lanes0-15 K=[0..7],[16..23]; lanes16-31 +8)
  const int bof = 8 * lh;

  const __bf16* ah[4]; const __bf16* al[4];
#pragma unroll
  for (int r = 0; r < 4; ++r) {
    const size_t off = (size_t)(rowBase + r * 16 + ln) * DIM;
    ah[r] = H + off; al[r] = L + off;
  }
  const __bf16* bh[2]; const __bf16* bl[2];
#pragma unroll
  for (int c = 0; c < 2; ++c) {
    const size_t off = (size_t)(colBase + c * 16 + ln) * DIM;
    bh[c] = H + off; bl[c] = L + off;
  }

  v8f acc[4][2];
  const v8f zero = {0.f, 0.f, 0.f, 0.f, 0.f, 0.f, 0.f, 0.f};
#pragma unroll
  for (int r = 0; r < 4; ++r)
#pragma unroll
    for (int c = 0; c < 2; ++c) acc[r][c] = zero;

  for (int k = 0; k < DIM; k += 32) {
    v16bf aH[4], aL[4], bH[2], bL[2];
#pragma unroll
    for (int r = 0; r < 4; ++r) {
      v8bf h0 = *(const v8bf*)(ah[r] + k + aof0);
      v8bf h1 = *(const v8bf*)(ah[r] + k + aof1);
      aH[r] = __builtin_shufflevector(h0, h1, 0, 1, 2, 3, 4, 5, 6, 7,
                                      8, 9, 10, 11, 12, 13, 14, 15);
      v8bf l0 = *(const v8bf*)(al[r] + k + aof0);
      v8bf l1 = *(const v8bf*)(al[r] + k + aof1);
      aL[r] = __builtin_shufflevector(l0, l1, 0, 1, 2, 3, 4, 5, 6, 7,
                                      8, 9, 10, 11, 12, 13, 14, 15);
    }
#pragma unroll
    for (int c = 0; c < 2; ++c) {
      v8bf q0 = *(const v8bf*)(bh[c] + k + bof);
      v8bf q1 = *(const v8bf*)(bh[c] + k + 16 + bof);
      bH[c] = __builtin_shufflevector(q0, q1, 0, 1, 2, 3, 4, 5, 6, 7,
                                      8, 9, 10, 11, 12, 13, 14, 15);
      v8bf p0 = *(const v8bf*)(bl[c] + k + bof);
      v8bf p1 = *(const v8bf*)(bl[c] + k + 16 + bof);
      bL[c] = __builtin_shufflevector(p0, p1, 0, 1, 2, 3, 4, 5, 6, 7,
                                      8, 9, 10, 11, 12, 13, 14, 15);
    }
#pragma unroll
    for (int r = 0; r < 4; ++r)
#pragma unroll
      for (int c = 0; c < 2; ++c) {
        acc[r][c] = __builtin_amdgcn_wmma_f32_16x16x32_bf16(
            false, aH[r], false, bH[c], (short)0, acc[r][c], false, false);
        acc[r][c] = __builtin_amdgcn_wmma_f32_16x16x32_bf16(
            false, aH[r], false, bL[c], (short)0, acc[r][c], false, false);
        acc[r][c] = __builtin_amdgcn_wmma_f32_16x16x32_bf16(
            false, aL[r], false, bH[c], (short)0, acc[r][c], false, false);
      }
  }

  const float* sq = ws + WS_SQ;
  const float ib0 = ws[WS_SCAL + 0], ib1 = ws[WS_SCAL + 1], ib2 = ws[WS_SCAL + 2],
              ib3 = ws[WS_SCAL + 3], ib4 = ws[WS_SCAL + 4];
  const float sgn = ((rowBase < BHALF) == (colBase < BHALF)) ? 1.f : -1.f;

  float sqj[2];
#pragma unroll
  for (int c = 0; c < 2; ++c) sqj[c] = sq[colBase + c * 16 + ln];

  float local = 0.f;
#pragma unroll
  for (int r = 0; r < 4; ++r) {
#pragma unroll
    for (int v = 0; v < 8; ++v) {
      const float sqi = sq[rowBase + r * 16 + v + 8 * lh];
#pragma unroll
      for (int c = 0; c < 2; ++c) {
        const float g = acc[r][c][v];
        float l2 = fmaf(-2.f, g, sqi + sqj[c]);
        l2 = fmaxf(l2, 0.f);
        local += __expf(-l2 * ib0) + __expf(-l2 * ib1) + __expf(-l2 * ib2) +
                 __expf(-l2 * ib3) + __expf(-l2 * ib4);
      }
    }
  }
  local = waveReduce(local);
  if (lane == 0) ws[WS_PART + blockIdx.x * 2 + wave] = wgt * sgn * local;
}

// ---------------- fallback path: exact fp32 WMMA 16x16x4 ----------------

__global__ __launch_bounds__(32) void k_mmd_gemm_f32(const float* __restrict__ s,
                                                     const float* __restrict__ t,
                                                     float* __restrict__ ws) {
  const int lane = threadIdx.x;
  const int ln = lane & 15, lh = lane >> 4;

  int I, J;
  triDecode(blockIdx.x, MSB, I, J);
  const int rowBase = I * 64;
  const int colBase = J * 64;
  const float wgt = (I == J) ? 1.f : 2.f;

  const float* Ar = (rowBase < BHALF) ? (s + (size_t)rowBase * DIM)
                                      : (t + (size_t)(rowBase - BHALF) * DIM);
  const float* Bc = (colBase < BHALF) ? (s + (size_t)colBase * DIM)
                                      : (t + (size_t)(colBase - BHALF) * DIM);

  const float* ap[4]; const float* bp[4];
#pragma unroll
  for (int r = 0; r < 4; ++r) ap[r] = Ar + (size_t)(r * 16 + ln) * DIM + 2 * lh;
#pragma unroll
  for (int c = 0; c < 4; ++c) bp[c] = Bc + (size_t)(c * 16 + ln) * DIM + 2 * lh;

  v8f acc[4][4];
  const v8f zero = {0.f, 0.f, 0.f, 0.f, 0.f, 0.f, 0.f, 0.f};
#pragma unroll
  for (int r = 0; r < 4; ++r)
#pragma unroll
    for (int c = 0; c < 4; ++c) acc[r][c] = zero;

  for (int k = 0; k < DIM; k += 4) {
    v2f a[4], b[4];
#pragma unroll
    for (int r = 0; r < 4; ++r) a[r] = *(const v2f*)(ap[r] + k);
#pragma unroll
    for (int c = 0; c < 4; ++c) b[c] = *(const v2f*)(bp[c] + k);
#pragma unroll
    for (int r = 0; r < 4; ++r)
#pragma unroll
      for (int c = 0; c < 4; ++c)
        acc[r][c] = __builtin_amdgcn_wmma_f32_16x16x4_f32(
            false, a[r], false, b[c], (short)0, acc[r][c], false, false);
  }

  const float* sq = ws + WS_SQ;
  const float ib0 = ws[WS_SCAL + 0], ib1 = ws[WS_SCAL + 1], ib2 = ws[WS_SCAL + 2],
              ib3 = ws[WS_SCAL + 3], ib4 = ws[WS_SCAL + 4];
  const float sgn = ((rowBase < BHALF) == (colBase < BHALF)) ? 1.f : -1.f;

  float sqj[4];
#pragma unroll
  for (int c = 0; c < 4; ++c) sqj[c] = sq[colBase + c * 16 + ln];

  float local = 0.f;
#pragma unroll
  for (int r = 0; r < 4; ++r) {
#pragma unroll
    for (int v = 0; v < 8; ++v) {
      const float sqi = sq[rowBase + r * 16 + v + 8 * lh];
#pragma unroll
      for (int c = 0; c < 4; ++c) {
        const float g = acc[r][c][v];
        float l2 = fmaf(-2.f, g, sqi + sqj[c]);
        l2 = fmaxf(l2, 0.f);
        local += __expf(-l2 * ib0) + __expf(-l2 * ib1) + __expf(-l2 * ib2) +
                 __expf(-l2 * ib3) + __expf(-l2 * ib4);
      }
    }
  }
  local = waveReduce(local);
  if (lane == 0) ws[WS_PART + blockIdx.x] = wgt * sgn * local;
}

// ---------------- final reduction ----------------

__global__ __launch_bounds__(256) void k_finalize(const float* __restrict__ ws,
                                                  float* __restrict__ out, int n) {
  __shared__ float sm[256];
  const int t = threadIdx.x;
  float acc = 0.f;
  for (int i = t; i < n; i += 256) acc += ws[WS_PART + i];
  sm[t] = acc; __syncthreads();
  for (int o = 128; o > 0; o >>= 1) { if (t < o) sm[t] += sm[t + o]; __syncthreads(); }
  if (t == 0) out[0] = sm[0] / ((float)BHALF * (float)BHALF);
}

extern "C" void kernel_launch(void* const* d_in, const int* in_sizes, int n_in,
                              void* d_out, int out_size, void* d_ws, size_t ws_size,
                              hipStream_t stream) {
  const float* src = (const float*)d_in[0];
  const float* tgt = (const float*)d_in[1];
  float* ws = (float*)d_ws;
  float* out = (float*)d_out;

  k_row_norms<<<NTOT / 8, 256, 0, stream>>>(src, tgt, ws);
  k_colpart<<<32, 256, 0, stream>>>(src, tgt, ws);
  k_scalars<<<1, 256, 0, stream>>>(ws);

  if (ws_size >= WS_NEED) {
    // fast path: split-bf16 + triangular symmetry (~5.3x fp32-WMMA cycles saved)
    __bf16* hiA = (__bf16*)((char*)d_ws + WS_HI_BYTES);
    __bf16* loA = (__bf16*)((char*)d_ws + WS_LO_BYTES);
    k_convert<<<(NTOT * DIM) / (256 * 4), 256, 0, stream>>>(src, tgt, hiA, loA);
    k_mmd_gemm_bf16<<<NSB, 64, 0, stream>>>(hiA, loA, ws);
    k_finalize<<<1, 256, 0, stream>>>(ws, out, NSB * 2);
  } else {
    k_mmd_gemm_f32<<<NSB, 32, 0, stream>>>(src, tgt, ws);
    k_finalize<<<1, 256, 0, stream>>>(ws, out, NSB);
  }
}
